// SparifyFFN_63153199121074
// MI455X (gfx1250) — matrix operand order repaired and checked by reading, not compiled, not run on hardware
//
#include <hip/hip_runtime.h>

// ---------------- problem constants (from reference) ----------------
// x: [B=4, S=2048, D=2048] f32 ; W1: [F=8192, D] ; Wm: [G=8, D] ; W2: [D, F]
// K(top-k)=1 -> one-hot group gate, group width GS = F/G = 1024.
namespace cfg {
constexpr int B = 4, S = 2048, D = 2048, F = 8192, G = 8;
constexpr int T   = B * S;         // 8192 tokens
constexpr int GS  = F / G;         // 1024 hidden cols per group
constexpr int M   = 32;            // tokens per FFN tile (2 WMMA M-tiles)
constexpr int CH  = 512;           // K-chunk for X staging
constexpr int XPc = CH + 8;        // LDS pitch (ushort) for x chunk
constexpr int HP  = GS + 8;        // LDS pitch (ushort) for hidden tile
}

typedef __bf16 v16bf __attribute__((ext_vector_type(16)));
typedef float  v8f   __attribute__((ext_vector_type(8)));

__device__ __forceinline__ unsigned short f2bf(float f) {
  unsigned u = __builtin_bit_cast(unsigned, f);
  unsigned r = u + 0x7FFFu + ((u >> 16) & 1u);   // round-to-nearest-even
  return (unsigned short)(r >> 16);
}

__device__ __forceinline__ v8f wmma_bf16(v16bf a, v16bf b, v8f c) {
  // D = A(16x32 bf16) * B(32x16 bf16) + C(16x16 f32)
  return __builtin_amdgcn_wmma_f32_16x16x32_bf16(
      /*neg_a=*/false, a, /*neg_b=*/false, b,
      /*c_mod=*/(short)0, c, /*reuse_a=*/false, /*reuse_b=*/false);
}

// A-fragment (16-bit 16x32): lane holds row (lane&15); elements e=0..15 -> K=k0+16*half+e.
// `p` is pre-offset by row*pitch + 16*half.
__device__ __forceinline__ v16bf ldA(const unsigned short* p, int k0) {
  union { struct { uint4 a, b; } u; v16bf v; } f;
  f.u.a = *reinterpret_cast<const uint4*>(p + k0);
  f.u.b = *reinterpret_cast<const uint4*>(p + k0 + 8);
  return f.v;
}

// B-fragment (16-bit 32x16): lane holds col (lane&15); elements e=0..15 -> K=k0+16*half+e.
// `p` pre-offset by row*pitch + 16*half.  (bf16 source)
__device__ __forceinline__ v16bf ldB_bf(const unsigned short* p, int k0) {
  union { struct { uint4 a, b; } u; v16bf v; } f;
  f.u.a = *reinterpret_cast<const uint4*>(p + k0);
  f.u.b = *reinterpret_cast<const uint4*>(p + k0 + 8);
  return f.v;
}

// Same, converting from f32 weights on the fly.
__device__ __forceinline__ v16bf ldB_f32(const float* p, int k0) {
  union { unsigned u[8]; v16bf v; } f;
  const float4* q = reinterpret_cast<const float4*>(p + k0);
#pragma unroll
  for (int i = 0; i < 4; ++i) {
    float4 t = q[i];
    f.u[2 * i + 0] = (unsigned)f2bf(t.x) | ((unsigned)f2bf(t.y) << 16);
    f.u[2 * i + 1] = (unsigned)f2bf(t.z) | ((unsigned)f2bf(t.w) << 16);
  }
  return f.v;
}

// ---------------- kernel 0: zero group counters ----------------
__global__ void zero_counts(int* gcount) {
  if (threadIdx.x < cfg::G) gcount[threadIdx.x] = 0;
}

// ---------------- kernel 1: gate (argmax group) + bucket tokens ----------------
// one wave32 per token; Wm (8x2048 f32 = 64KB) staged in LDS.
__global__ __launch_bounds__(256) void gate_kernel(
    const float* __restrict__ x, const float* __restrict__ Wm,
    const float* __restrict__ bm, int* __restrict__ gcount,
    int* __restrict__ glist) {
  extern __shared__ float sWm[];  // G*D floats
  const int tid = threadIdx.x;
  for (int i = tid; i < cfg::G * cfg::D; i += 256) sWm[i] = Wm[i];
  __syncthreads();

  const int wave = tid >> 5, lane = tid & 31;
  const int t = blockIdx.x * 8 + wave;
  const float* xr = x + (size_t)t * cfg::D;

  float acc[cfg::G];
#pragma unroll
  for (int g = 0; g < cfg::G; ++g) acc[g] = 0.f;

  for (int d = lane; d < cfg::D; d += 32) {
    float xv = xr[d];
#pragma unroll
    for (int g = 0; g < cfg::G; ++g) acc[g] = fmaf(xv, sWm[g * cfg::D + d], acc[g]);
  }
#pragma unroll
  for (int g = 0; g < cfg::G; ++g) {
#pragma unroll
    for (int off = 16; off > 0; off >>= 1) acc[g] += __shfl_xor(acc[g], off, 32);
  }
  if (lane == 0) {
    // softmax is monotonic; top-1 == argmax of logits; strict '>' keeps lowest
    // index on ties (matches jax.lax.top_k stability).
    float best = acc[0] + bm[0];
    int bg = 0;
#pragma unroll
    for (int g = 1; g < cfg::G; ++g) {
      float v = acc[g] + bm[g];
      if (v > best) { best = v; bg = g; }
    }
    int pos = atomicAdd(&gcount[bg], 1);
    glist[bg * cfg::T + pos] = t;
  }
}

// ---------------- kernel 2: one-shot weight f32 -> bf16 conversion ----------------
__global__ void cvt_weights(const float* __restrict__ W1, const float* __restrict__ W2,
                            unsigned short* __restrict__ W1b,
                            unsigned short* __restrict__ W2b) {
  const size_t n = (size_t)cfg::F * cfg::D;  // same count for both
  for (size_t i = (size_t)blockIdx.x * blockDim.x + threadIdx.x; i < n;
       i += (size_t)gridDim.x * blockDim.x) {
    W1b[i] = f2bf(W1[i]);
    W2b[i] = f2bf(W2[i]);
  }
}

// ---------------- kernel 3: sparse FFN via bf16 WMMA ----------------
// block = 256 threads (8 waves). blockIdx.y = group, blockIdx.x = 32-token tile
// within that group's bucket. Phase A: H = relu(X*W1_g^T + b1) kept as bf16 in
// LDS (X staged in K=512 chunks to keep LDS ~99KB -> ~3 blocks/WGP).
// Phase B: OUT = H*W2_g^T + b2.
template <bool PRE>
__global__ __launch_bounds__(256) void ffn_kernel(
    const float* __restrict__ x, const float* __restrict__ W1f,
    const float* __restrict__ b1, const float* __restrict__ W2f,
    const float* __restrict__ b2, const unsigned short* __restrict__ W1b,
    const unsigned short* __restrict__ W2b, const int* __restrict__ gcount,
    const int* __restrict__ glist, float* __restrict__ out) {
  using namespace cfg;
  extern __shared__ unsigned char smem[];
  unsigned short* sX = (unsigned short*)smem;                           // [M][XPc]
  unsigned short* sH = (unsigned short*)(smem + (size_t)M * XPc * 2);   // [M][HP]
  int* sTok = (int*)(smem + (size_t)M * XPc * 2 + (size_t)M * HP * 2);  // [M]

  const int g = blockIdx.y;
  const int nt = gcount[g];
  const int tile = blockIdx.x;
  if (tile * M >= nt) return;  // uniform per block
  const int gbase = g * GS;

  const int tid = threadIdx.x;
  if (tid < M) {
    int idx = tile * M + tid;
    if (idx >= nt) idx = nt - 1;  // duplicate last token; writes identical data
    sTok[tid] = glist[g * T + idx];
  }
  __syncthreads();

  const int wave = tid >> 5, lane = tid & 31;
  const int l15 = lane & 15, half = lane >> 4;

  // ---------------- Phase A: H[32][1024] ----------------
  {
    v8f acc[2][8];
#pragma unroll
    for (int mt = 0; mt < 2; ++mt)
#pragma unroll
      for (int j = 0; j < 8; ++j) acc[mt][j] = v8f{};

    const unsigned short* pBb[8];
    const float* pBf[8];
#pragma unroll
    for (int j = 0; j < 8; ++j) {
      size_t row = (size_t)(gbase + wave * 128 + j * 16 + l15);
      if constexpr (PRE) pBb[j] = W1b + row * D + 16 * half;
      else               pBf[j] = W1f + row * D + 16 * half;
    }

    const unsigned short* a0 = sX + (0 * 16 + l15) * XPc + 16 * half;
    const unsigned short* a1 = sX + (1 * 16 + l15) * XPc + 16 * half;

    for (int ck = 0; ck < D / CH; ++ck) {
      __syncthreads();  // previous chunk fully consumed before overwrite
      for (int i = tid; i < M * CH; i += 256) {
        int m = i >> 9, d = i & (CH - 1);
        sX[m * XPc + d] = f2bf(x[(size_t)sTok[m] * D + ck * CH + d]);
      }
      __syncthreads();

      // pull next chunk's weight lines toward the WGP while this chunk computes
      if (ck + 1 < D / CH) {
#pragma unroll
        for (int j = 0; j < 8; ++j) {
          if constexpr (PRE) __builtin_prefetch(pBb[j] + (ck + 1) * CH, 0, 3);
          else               __builtin_prefetch(pBf[j] + (ck + 1) * CH, 0, 3);
        }
      }

      for (int k0 = 0; k0 < CH; k0 += 32) {
        const int kk = ck * CH + k0;
        v16bf fa0 = ldA(a0, k0), fa1 = ldA(a1, k0);
#pragma unroll
        for (int j = 0; j < 8; ++j) {
          v16bf fb;
          if constexpr (PRE) fb = ldB_bf(pBb[j], kk);
          else               fb = ldB_f32(pBf[j], kk);
          acc[0][j] = wmma_bf16(fa0, fb, acc[0][j]);
          acc[1][j] = wmma_bf16(fa1, fb, acc[1][j]);
        }
      }
    }

    // epilogue: +b1, relu, write bf16 hidden tile
#pragma unroll
    for (int mt = 0; mt < 2; ++mt)
#pragma unroll
      for (int j = 0; j < 8; ++j) {
        int col = wave * 128 + j * 16 + l15;
        float bias = b1[gbase + col];
#pragma unroll
        for (int r = 0; r < 8; ++r) {
          float v = acc[mt][j][r] + bias;
          v = v > 0.f ? v : 0.f;
          sH[(mt * 16 + half * 8 + r) * HP + col] = f2bf(v);
        }
      }
  }
  __syncthreads();

  // ---------------- Phase B: OUT[32][2048] ----------------
  int tokRow[2][8];
#pragma unroll
  for (int mt = 0; mt < 2; ++mt)
#pragma unroll
    for (int r = 0; r < 8; ++r) tokRow[mt][r] = sTok[mt * 16 + half * 8 + r];

  for (int nh = 0; nh < 2; ++nh) {
    v8f acc[2][8];
#pragma unroll
    for (int mt = 0; mt < 2; ++mt)
#pragma unroll
      for (int j = 0; j < 8; ++j) acc[mt][j] = v8f{};

    const unsigned short* a0 = sH + (0 * 16 + l15) * HP + 16 * half;
    const unsigned short* a1 = sH + (1 * 16 + l15) * HP + 16 * half;

    const unsigned short* pBb[8];
    const float* pBf[8];
#pragma unroll
    for (int j = 0; j < 8; ++j) {
      size_t row = (size_t)(wave * 256 + nh * 128 + j * 16 + l15);  // out col d
      if constexpr (PRE) pBb[j] = W2b + row * F + gbase + 16 * half;
      else               pBf[j] = W2f + row * F + gbase + 16 * half;
    }

    for (int k0 = 0; k0 < GS; k0 += 32) {
      v16bf fa0 = ldA(a0, k0), fa1 = ldA(a1, k0);
#pragma unroll
      for (int j = 0; j < 8; ++j) {
        v16bf fb;
        if constexpr (PRE) fb = ldB_bf(pBb[j], k0);
        else               fb = ldB_f32(pBf[j], k0);
        acc[0][j] = wmma_bf16(fa0, fb, acc[0][j]);
        acc[1][j] = wmma_bf16(fa1, fb, acc[1][j]);
      }
    }

#pragma unroll
    for (int mt = 0; mt < 2; ++mt)
#pragma unroll
      for (int j = 0; j < 8; ++j) {
        int col = wave * 256 + nh * 128 + j * 16 + l15;
        float bias = b2[col];
#pragma unroll
        for (int r = 0; r < 8; ++r)
          out[(size_t)tokRow[mt][r] * D + col] = acc[mt][j][r] + bias;
      }
  }
}

// ---------------- host launcher ----------------
extern "C" void kernel_launch(void* const* d_in, const int* in_sizes, int n_in,
                              void* d_out, int out_size, void* d_ws, size_t ws_size,
                              hipStream_t stream) {
  (void)in_sizes; (void)n_in; (void)out_size;
  using namespace cfg;
  const float* x  = (const float*)d_in[0];
  const float* W1 = (const float*)d_in[1];
  const float* b1 = (const float*)d_in[2];
  const float* Wm = (const float*)d_in[3];
  const float* bm = (const float*)d_in[4];
  const float* W2 = (const float*)d_in[5];
  const float* b2 = (const float*)d_in[6];
  float* out = (float*)d_out;

  const size_t wBytes = (size_t)F * D * sizeof(unsigned short);  // 32 MB each
  const size_t listBytes = (size_t)G * T * sizeof(int);
  const size_t needFull = 2 * wBytes + 1024 + listBytes;
  const bool pre = ws_size >= needFull;

  unsigned char* ws = (unsigned char*)d_ws;
  unsigned short* W1b = nullptr;
  unsigned short* W2b = nullptr;
  int* gcount;
  int* glist;
  if (pre) {
    W1b = (unsigned short*)ws;
    W2b = (unsigned short*)(ws + wBytes);
    gcount = (int*)(ws + 2 * wBytes);
    glist = (int*)(ws + 2 * wBytes + 1024);
  } else {
    gcount = (int*)ws;
    glist = (int*)(ws + 1024);
  }

  zero_counts<<<1, 32, 0, stream>>>(gcount);
  gate_kernel<<<T / 8, 256, G * D * sizeof(float), stream>>>(x, Wm, bm, gcount, glist);

  const size_t smem = (size_t)M * XPc * 2 + (size_t)M * HP * 2 + 128;  // ~99.5 KB
  dim3 grid(T / M, G);
  if (pre) {
    cvt_weights<<<4096, 256, 0, stream>>>(W1, W2, W1b, W2b);
    ffn_kernel<true><<<grid, 256, smem, stream>>>(x, W1, b1, W2, b2, W1b, W2b,
                                                  gcount, glist, out);
  } else {
    ffn_kernel<false><<<grid, 256, smem, stream>>>(x, W1, b1, W2, b2, W1b, W2b,
                                                   gcount, glist, out);
  }
}
